// PointTNT_66486093742750
// MI455X (gfx1250) — matrix operand order repaired and checked by reading, not compiled, not run on hardware
//
#include <hip/hip_runtime.h>
#include <math.h>

#define EPSF 1e-5f

typedef __attribute__((ext_vector_type(16))) _Float16 v16h;
typedef __attribute__((ext_vector_type(8)))  float    v8f;

// =====================================================================
// Generic batched WMMA GEMM.
// wave32: one wave computes a 16 x (16*NT) output tile, K stepped by 32.
// A fragment loaded once per K-step; all NT B fragments are loaded into
// distinct registers BEFORE the MAC phase so the loads issue as a clause.
// AT/BT = float (convert on load) or _Float16 (native, no cvt in loop).
// TB: B given row-major (N x K), computes A*B^T.
// dmode: 0 -> store D as fp32, 1 -> store D as f16 (same indexing).
// D[i0,i1][m,n] = act( alpha*sum_k A*B + bias[n] | bn-affine ) + R[m,n]
// =====================================================================
template<int NT, bool TB, typename AT, typename BT>
__global__ void __launch_bounds__(128)
k_gemm(const AT* __restrict__ A, const BT* __restrict__ Bm,
       const float* __restrict__ bias,
       const float* __restrict__ bnG, const float* __restrict__ bnB,
       const float* __restrict__ R, float* __restrict__ D,
       int dmode, int M, int Kd, int Nd,
       int lda, int ldb, int ldd, int ldr,
       long sA0, long sA1, long sB0, long sB1,
       long sD0, long sD1, long sR0, long sR1,
       int nb1, int tilesM, int tilesN, long totalWaves,
       int act, float alpha)
{
  long gw = (long)blockIdx.x * 4 + (threadIdx.x >> 5);
  if (gw >= totalWaves) return;                 // wave-uniform: EXEC stays full
  int lane = threadIdx.x & 31;

  int  tn = (int)(gw % tilesN);
  long t1 = gw / tilesN;
  int  tm = (int)(t1 % tilesM);
  long t2 = t1 / tilesM;
  int  i1 = (int)(t2 % nb1);
  int  i0 = (int)(t2 / nb1);

  const AT* Ab = A  + (long)i0 * sA0 + (long)i1 * sA1;
  const BT* Bb = Bm + (long)i0 * sB0 + (long)i1 * sB1;

  int mrow  = tm * 16 + (lane & 15);
  int mload = mrow < M ? mrow : M - 1;
  int kbA   = (lane >> 4) << 4;                 // A 16x32 layout: K half 0 or 16 (matches B)
  int kbB   = (lane >> 4) << 4;                 // B 32x16 layout: K half 0 or 16
  int nbase = tn * (16 * NT) + (lane & 15);

  int nld[NT];
#pragma unroll
  for (int j = 0; j < NT; ++j) {
    int nc = nbase + 16 * j;
    nld[j] = nc < Nd ? nc : Nd - 1;
  }

  const AT* arow = Ab + (long)mload * lda;
  v8f zero = {0.f, 0.f, 0.f, 0.f, 0.f, 0.f, 0.f, 0.f};
  v8f acc[NT];
#pragma unroll
  for (int j = 0; j < NT; ++j) acc[j] = zero;

  for (int kk = 0; kk < Kd; kk += 32) {
    v16h af;
#pragma unroll
    for (int e = 0; e < 8; ++e) {
      af[e]     = (_Float16)arow[kk + kbA + e];
      af[e + 8] = (_Float16)arow[kk + kbA + 8 + e];
    }
    v16h bf[NT];                                 // distinct regs: loads overlap
#pragma unroll
    for (int j = 0; j < NT; ++j) {
      if (TB) {
        const BT* bp = Bb + (long)nld[j] * ldb + kk + kbB;
#pragma unroll
        for (int e = 0; e < 16; ++e) bf[j][e] = (_Float16)bp[e];
      } else {
        const BT* bp = Bb + (long)(kk + kbB) * ldb + nld[j];
#pragma unroll
        for (int e = 0; e < 16; ++e) bf[j][e] = (_Float16)bp[(long)e * ldb];
      }
    }
#pragma unroll
    for (int j = 0; j < NT; ++j)
      acc[j] = __builtin_amdgcn_wmma_f32_16x16x32_f16(false, af, false, bf[j],
                                                      (short)0, acc[j], false, false);
  }

  long  dbase   = (long)i0 * sD0 + (long)i1 * sD1;
  long  rbase   = (long)i0 * sR0 + (long)i1 * sR1;
  float bnscale = rsqrtf(1.f + EPSF);
#pragma unroll
  for (int j = 0; j < NT; ++j) {
    int nc = nbase + 16 * j;
    if (nc < Nd) {
#pragma unroll
      for (int r = 0; r < 8; ++r) {
        int mo = tm * 16 + r + ((lane >> 4) << 3);  // C/D: M = r + 8*(lane/16)
        if (mo < M) {
          float v = acc[j][r] * alpha;
          if (bias) v += bias[nc];
          if (bnG)  v = v * bnscale * bnG[nc] + bnB[nc];  // eval-mode BN
          if (act == 1) v = 0.5f * v * (1.f + erff(v * 0.70710678118654752440f));
          if (R) v += R[rbase + (long)mo * ldr + nc];
          long di = dbase + (long)mo * ldd + nc;
          if (dmode) ((_Float16*)D)[di] = (_Float16)v;
          else       D[di] = v;
        }
      }
    }
  }
}

// ------- weight pre-pack: W (K x N, fp32) -> Wp (N x K, f16, transposed) ------
__global__ void k_packT(const float* __restrict__ W, _Float16* __restrict__ Wp,
                        int Kd, int Nd)
{
  long id = (long)blockIdx.x * 256 + threadIdx.x;
  if (id >= (long)Kd * Nd) return;
  int n = (int)(id / Kd), k = (int)(id % Kd);
  Wp[id] = (_Float16)W[(long)k * Nd + n];
}

// ---------------- LayerNorm (one wave / row), f16 or f32 output ---------------
__global__ void __launch_bounds__(128)
k_layernorm(const float* __restrict__ X, float* __restrict__ Y,
            const float* __restrict__ g, const float* __restrict__ b,
            long rows, int Dd, int ldx, int ldy, int of16)
{
  long row = (long)blockIdx.x * 4 + (threadIdx.x >> 5);
  if (row >= rows) return;
  int lane = threadIdx.x & 31;
  const float* xp = X + row * (long)ldx;
  float s = 0.f, ss = 0.f;
  for (int j = lane; j < Dd; j += 32) { float v = xp[j]; s += v; ss += v * v; }
  for (int o = 16; o; o >>= 1) { s += __shfl_xor(s, o, 32); ss += __shfl_xor(ss, o, 32); }
  float mean = s / Dd;
  float inv  = rsqrtf(ss / Dd - mean * mean + EPSF);
  if (of16) {
    _Float16* yp = (_Float16*)Y + row * (long)ldy;
    for (int j = lane; j < Dd; j += 32) yp[j] = (_Float16)((xp[j] - mean) * inv * g[j] + b[j]);
  } else {
    float* yp = Y + row * (long)ldy;
    for (int j = lane; j < Dd; j += 32) yp[j] = (xp[j] - mean) * inv * g[j] + b[j];
  }
}

// ------------------------- row softmax (one wave / row) -----------------------
__global__ void __launch_bounds__(128)
k_softmax(float* __restrict__ S, long rows, int n)
{
  long row = (long)blockIdx.x * 4 + (threadIdx.x >> 5);
  if (row >= rows) return;
  int lane = threadIdx.x & 31;
  float* p = S + row * (long)n;
  float mx = -1e30f;
  for (int j = lane; j < n; j += 32) mx = fmaxf(mx, p[j]);
  for (int o = 16; o; o >>= 1) mx = fmaxf(mx, __shfl_xor(mx, o, 32));
  float sum = 0.f;
  for (int j = lane; j < n; j += 32) { float e = expf(p[j] - mx); p[j] = e; sum += e; }
  for (int o = 16; o; o >>= 1) sum += __shfl_xor(sum, o, 32);
  float inv = 1.f / sum;
  for (int j = lane; j < n; j += 32) p[j] *= inv;
}

// --------- per-sequence [max ; mean] pooling over tokens -> (nseq, 2D) --------
__global__ void k_maxmean(const float* __restrict__ X, float* __restrict__ Y,
                          int ntok, int Dd, int ldx)
{
  int s = blockIdx.x;
  const float* base = X + (long)s * ntok * ldx;
  for (int d = threadIdx.x; d < Dd; d += blockDim.x) {
    float mx = -1e30f, sm = 0.f;
    for (int t = 0; t < ntok; ++t) {
      float v = base[(long)t * ldx + d];
      mx = fmaxf(mx, v); sm += v;
    }
    Y[(long)s * 2 * Dd + d]      = mx;
    Y[(long)s * 2 * Dd + Dd + d] = sm / ntok;
  }
}

// ------------------------- (B,3,N) -> (B,N,3) transpose -----------------------
__global__ void k_xpose(const float* __restrict__ x, float* __restrict__ xyz,
                        int Bn, int Np)
{
  int idx = blockIdx.x * blockDim.x + threadIdx.x;
  if (idx >= Bn * Np) return;
  int b = idx / Np, n = idx % Np;
  for (int c = 0; c < 3; ++c)
    xyz[(long)idx * 3 + c] = x[((long)b * 3 + c) * Np + n];
}

// ------------- deterministic farthest point sampling (1 block / batch) --------
__global__ void __launch_bounds__(256)
k_fps(const float* __restrict__ xyz, int* __restrict__ fidx, float* __restrict__ anch)
{
  __shared__ float dist[2048];
  __shared__ float rv[256];
  __shared__ int   ri[256];
  __shared__ int   sfar;
  int b = blockIdx.x, tid = threadIdx.x;
  const float* pb = xyz + (long)b * 2048 * 3;
  for (int n = tid; n < 2048; n += 256) dist[n] = 1e10f;
  if (tid == 0) sfar = 0;
  __syncthreads();
  for (int it = 0; it < 256; ++it) {
    int far = sfar;
    float cx = pb[far * 3], cy = pb[far * 3 + 1], cz = pb[far * 3 + 2];
    if (tid == 0) {
      fidx[b * 256 + it] = far;
      anch[((long)b * 256 + it) * 3 + 0] = cx;
      anch[((long)b * 256 + it) * 3 + 1] = cy;
      anch[((long)b * 256 + it) * 3 + 2] = cz;
    }
    float best = -1.f; int bi = 2048;
    for (int n = tid; n < 2048; n += 256) {
      float dx = pb[n * 3] - cx, dy = pb[n * 3 + 1] - cy, dz = pb[n * 3 + 2] - cz;
      float d = fminf(dist[n], dx * dx + dy * dy + dz * dz);
      dist[n] = d;
      if (d > best || (d == best && n < bi)) { best = d; bi = n; }
    }
    rv[tid] = best; ri[tid] = bi;
    __syncthreads();
    for (int s2 = 128; s2; s2 >>= 1) {
      if (tid < s2) {
        if (rv[tid + s2] > rv[tid] ||
            (rv[tid + s2] == rv[tid] && ri[tid + s2] < ri[tid])) {
          rv[tid] = rv[tid + s2]; ri[tid] = ri[tid + s2];
        }
      }
      __syncthreads();
    }
    if (tid == 0) sfar = ri[0];
    __syncthreads();
  }
}

// ----------------- kNN(32) for each anchor (1 block / anchor) -----------------
__global__ void __launch_bounds__(256)
k_knn(const float* __restrict__ xyz, const float* __restrict__ anch,
      int* __restrict__ aknn)
{
  __shared__ float dist[2048];
  __shared__ float rv[256];
  __shared__ int   ri[256];
  int t = blockIdx.x;               // b*256 + p
  int b = t >> 8;
  int tid = threadIdx.x;
  const float* pb = xyz + (long)b * 2048 * 3;
  float ax = anch[t * 3], ay = anch[t * 3 + 1], az = anch[t * 3 + 2];
  for (int n = tid; n < 2048; n += 256) {
    float dx = pb[n * 3] - ax, dy = pb[n * 3 + 1] - ay, dz = pb[n * 3 + 2] - az;
    dist[n] = dx * dx + dy * dy + dz * dz;
  }
  __syncthreads();
  for (int kth = 0; kth < 32; ++kth) {
    float best = 1e30f; int bi = 2048;
    for (int n = tid; n < 2048; n += 256) {
      float d = dist[n];
      if (d < best || (d == best && n < bi)) { best = d; bi = n; }
    }
    rv[tid] = best; ri[tid] = bi;
    __syncthreads();
    for (int s2 = 128; s2; s2 >>= 1) {
      if (tid < s2) {
        if (rv[tid + s2] < rv[tid] ||
            (rv[tid + s2] == rv[tid] && ri[tid + s2] < ri[tid])) {
          rv[tid] = rv[tid + s2]; ri[tid] = ri[tid + s2];
        }
      }
      __syncthreads();
    }
    if (tid == 0) { aknn[(long)t * 32 + kth] = ri[0]; dist[ri[0]] = 1e30f; }
    __syncthreads();
  }
}

// --------- edge features (neigh - anchor) @ Wpt(3,64) + bpt  (K=3: scalar) ----
__global__ void k_edge(const float* __restrict__ xyz, const float* __restrict__ anch,
                       const int* __restrict__ aknn, const float* __restrict__ Wpt,
                       const float* __restrict__ bpt, float* __restrict__ edges)
{
  long tok = blockIdx.x;            // B*P*K tokens
  int  o   = threadIdx.x;           // 64 out dims
  int  b   = (int)(tok >> 13);      // P*K = 8192
  long ap  = tok >> 5;              // b*P + p
  int  nn  = aknn[tok];
  float e0 = xyz[((long)b * 2048 + nn) * 3 + 0] - anch[ap * 3 + 0];
  float e1 = xyz[((long)b * 2048 + nn) * 3 + 1] - anch[ap * 3 + 1];
  float e2 = xyz[((long)b * 2048 + nn) * 3 + 2] - anch[ap * 3 + 2];
  edges[tok * 64 + o] = bpt[o] + e0 * Wpt[o] + e1 * Wpt[64 + o] + e2 * Wpt[128 + o];
}

// ------------------- anchors @ Wan(3,256) + ban  (K=3: scalar) ----------------
__global__ void k_anchor(const float* __restrict__ anch, const float* __restrict__ Wan,
                         const float* __restrict__ ban, float* __restrict__ p0)
{
  long t = blockIdx.x; int o = threadIdx.x;
  p0[t * 256 + o] = ban[o] + anch[t * 3] * Wan[o] + anch[t * 3 + 1] * Wan[256 + o] +
                    anch[t * 3 + 2] * Wan[512 + o];
}

// =====================================================================
extern "C" void kernel_launch(void* const* d_in, const int* in_sizes, int n_in,
                              void* d_out, int out_size, void* d_ws, size_t ws_size,
                              hipStream_t stream)
{
  (void)out_size; (void)ws_size;
  constexpr int Bn = 16, Np = 2048, Pn = 256, Kn = 32, HEADS = 4;
  const float SCALE = 0.125f;      // 64^-0.5

  // ---- input unpack: 'x' then params flattened as a sorted-key pytree -------
  int xi = (in_sizes[0] == Bn * 3 * Np) ? 0 : (n_in - 1);
  int pbi = (xi == 0) ? 1 : 0;
  const float* X = (const float*)d_in[xi];
  auto PR = [&](int i) -> const float* { return (const float*)d_in[pbi + i]; };
  // 0:Wan 1:Wpt 2:ban 3:bpt 4:fc.W 5:fc.b 6:fc.lnb 7:fc.lng
  // 8:h.W1 9:h.W2 10:h.W3 11:h.b1 12:h.b2 13:h.b3 14:h.be1 15:h.be2 16:h.g1 17:h.g2
  // layer lp at 18+lp*26: gattn{Wo,Wqkv,bo,lnb,lng} gff{W1,W2,b1,b2,lnb,lng}
  //                       p2p{W,b,lnb,lng} pattn{Wo,Wqkv,bo,lnb,lng} pff{W1,W2,b1,b2,lnb,lng}
  const float *Wan = PR(0), *Wpt = PR(1), *ban = PR(2), *bpt = PR(3);
  const float *fcW = PR(4), *fcb = PR(5), *fclnb = PR(6), *fclng = PR(7);
  const float *hW1 = PR(8), *hW2 = PR(9), *hW3 = PR(10);
  const float *hb1 = PR(11), *hb2 = PR(12), *hb3 = PR(13);
  const float *hbe1 = PR(14), *hbe2 = PR(15), *hg1 = PR(16), *hg2 = PR(17);

  // ---- workspace arena (~146 MB fp32 + ~20 MB f16; L2-resident) -------------
  float* ws = (float*)d_ws; size_t off = 0;
  auto alloc = [&](size_t n) -> float* {
    float* p = ws + off; off += (n + 63) & ~(size_t)63; return p;
  };
  float* xyz   = alloc((size_t)Bn * Np * 3);
  float* anch  = alloc((size_t)Bn * Pn * 3);
  int*   fidx  = (int*)alloc((size_t)Bn * Pn);
  int*   aknn  = (int*)alloc((size_t)Bn * Pn * Kn);
  float* edges = alloc((size_t)Bn * Pn * Kn * 64);   // residual stream, point level
  float* bufA  = alloc((size_t)32768 * 256);         // fp32 / f16-aliased scratch
  float* bufB  = alloc((size_t)32768 * 256);
  float* Sbuf  = alloc((size_t)4194304);             // attention scores
  float* ef    = alloc((size_t)4096 * 128);
  float* pwork = alloc((size_t)4096 * 256);          // patch residual stream (this layer)
  float* p0    = alloc((size_t)4096 * 256);
  float* ybuf  = alloc((size_t)4096 * 1024);         // depth-concat of patches
  float* pooled= alloc((size_t)16 * 2048);
  float* h1b   = alloc((size_t)16 * 512);
  float* h2b   = alloc((size_t)16 * 256);
  // f16 arena: LN outputs + packed weights (256B aligned)
  _Float16* harena = (_Float16*)(ws + off);
  size_t hoff = 0;
  auto halloc = [&](size_t n) -> _Float16* {
    _Float16* p = harena + hoff; hoff += (n + 63) & ~(size_t)63; return p;
  };
  _Float16* lnh = halloc((size_t)4096 * 1024);       // f16 LayerNorm outputs
  auto PACK = [&](const float* W, int Kd, int Nd) -> const _Float16* {
    _Float16* p = halloc((size_t)Kd * Nd);
    long tot = (long)Kd * Nd;
    k_packT<<<(int)((tot + 255) / 256), 256, 0, stream>>>(W, p, Kd, Nd);
    return p;
  };
  _Float16* bufAh = (_Float16*)bufA;                 // f16 views of scratch
  _Float16* bufBh = (_Float16*)bufB;

  // ---- launch helpers -------------------------------------------------------
#define KG_TAIL nb1, tilesM, tilesN, tw, act, alpha
  // (f16 A) x (f16 B^T): all weight linears AND QK^T
  auto G_HH = [&](const _Float16* A, const _Float16* Bm, const float* bias,
                  const float* bnG, const float* bnB, const float* R, float* Dp,
                  int dmode, int M, int Kd, int Nd, int lda, int ldb, int ldd, int ldr,
                  long sA0, long sA1, long sB0, long sB1,
                  long sD0, long sD1, long sR0, long sR1,
                  int nb0, int nb1, int act, float alpha) {
    int nt = (Nd % 64 == 0) ? 4 : (Nd % 32 == 0) ? 2 : 1;
    int tilesM = (M + 15) / 16, tilesN = (Nd + 16 * nt - 1) / (16 * nt);
    long tw = (long)nb0 * nb1 * tilesM * tilesN;
    int blocks = (int)((tw + 3) / 4);
#define KG_ARGS A, Bm, bias, bnG, bnB, R, Dp, dmode, M, Kd, Nd, lda, ldb, ldd, ldr, \
                sA0, sA1, sB0, sB1, sD0, sD1, sR0, sR1, KG_TAIL
    if (nt == 4)      k_gemm<4, true, _Float16, _Float16><<<blocks, 128, 0, stream>>>(KG_ARGS);
    else if (nt == 2) k_gemm<2, true, _Float16, _Float16><<<blocks, 128, 0, stream>>>(KG_ARGS);
    else              k_gemm<1, true, _Float16, _Float16><<<blocks, 128, 0, stream>>>(KG_ARGS);
#undef KG_ARGS
  };
  // (f32 A) x (f16 B^T): head MLP on fp32 pooled activations
  auto G_FH = [&](const float* A, const _Float16* Bm, const float* bias,
                  const float* bnG, const float* bnB, float* Dp,
                  int M, int Kd, int Nd, int lda, int ldd, int act) {
    int nt = (Nd % 64 == 0) ? 4 : (Nd % 32 == 0) ? 2 : 1;
    int tilesM = (M + 15) / 16, tilesN = (Nd + 16 * nt - 1) / (16 * nt);
    long tw = (long)tilesM * tilesN;
    int blocks = (int)((tw + 3) / 4);
    int nb1 = 1; float alpha = 1.f;
#define KG_ARGS A, Bm, bias, bnG, bnB, (const float*)nullptr, Dp, 0, M, Kd, Nd, lda, Kd, ldd, 0, \
                (long)0, (long)0, (long)0, (long)0, (long)0, (long)0, (long)0, (long)0, KG_TAIL
    if (nt == 4)      k_gemm<4, true, float, _Float16><<<blocks, 128, 0, stream>>>(KG_ARGS);
    else if (nt == 2) k_gemm<2, true, float, _Float16><<<blocks, 128, 0, stream>>>(KG_ARGS);
    else              k_gemm<1, true, float, _Float16><<<blocks, 128, 0, stream>>>(KG_ARGS);
#undef KG_ARGS
  };
  // (f32 A) x (f32 B), NN: P·V (softmax probs x fp32 V)
  auto G_FF = [&](const float* A, const float* Bm, float* Dp, int dmode,
                  int M, int Kd, int Nd, int lda, int ldb, int ldd,
                  long sA0, long sA1, long sB0, long sB1, long sD0, long sD1,
                  int nb0, int nb1) {
    int nt = (Nd % 64 == 0) ? 4 : (Nd % 32 == 0) ? 2 : 1;
    int tilesM = (M + 15) / 16, tilesN = (Nd + 16 * nt - 1) / (16 * nt);
    long tw = (long)nb0 * nb1 * tilesM * tilesN;
    int blocks = (int)((tw + 3) / 4);
    int act = 0; float alpha = 1.f;
#define KG_ARGS A, Bm, (const float*)nullptr, (const float*)nullptr, (const float*)nullptr, \
                (const float*)nullptr, Dp, dmode, M, Kd, Nd, lda, ldb, ldd, 0, \
                sA0, sA1, sB0, sB1, sD0, sD1, (long)0, (long)0, KG_TAIL
    if (nt == 4)      k_gemm<4, false, float, float><<<blocks, 128, 0, stream>>>(KG_ARGS);
    else if (nt == 2) k_gemm<2, false, float, float><<<blocks, 128, 0, stream>>>(KG_ARGS);
    else              k_gemm<1, false, float, float><<<blocks, 128, 0, stream>>>(KG_ARGS);
#undef KG_ARGS
  };
  // flat (token-parallel) f16 weight linear, no batching
  auto LINH = [&](const _Float16* A, const _Float16* Wp, const float* bias,
                  const float* R, float* Dp, int dmode,
                  int M, int Kd, int Nd, int lda, int ldd, int ldr, int act) {
    G_HH(A, Wp, bias, nullptr, nullptr, R, Dp, dmode, M, Kd, Nd, lda, Kd, ldd, ldr,
         0, 0, 0, 0, 0, 0, 0, 0, 1, 1, act, 1.f);
  };
  auto LNRM = [&](const float* Xp, _Float16* Yp, const float* g, const float* b,
                  long rows, int Dd, int ldx, int ldy) {
    k_layernorm<<<(int)((rows + 3) / 4), 128, 0, stream>>>(Xp, (float*)Yp, g, b,
                                                           rows, Dd, ldx, ldy, 1);
  };
  auto SMAX = [&](float* S, long rows, int n) {
    k_softmax<<<(int)((rows + 3) / 4), 128, 0, stream>>>(S, rows, n);
  };

  // ---- pre-pack all weights to f16, transposed ------------------------------
  const _Float16* fcWp = PACK(fcW, 1024, 1024);
  const _Float16* hW1p = PACK(hW1, 2048, 512);
  const _Float16* hW2p = PACK(hW2, 512, 256);
  const _Float16* hW3p = PACK(hW3, 256, 40);
  const _Float16 *gWqkvP[4], *gWoP[4], *gW1P[4], *gW2P[4], *pWP[4],
                 *aWqkvP[4], *aWoP[4], *fW1P[4], *fW2P[4];
  for (int lp = 0; lp < 4; ++lp) {
    int L = 18 + lp * 26;
    gWoP[lp]   = PACK(PR(L + 0), 256, 256);
    gWqkvP[lp] = PACK(PR(L + 1), 256, 768);
    gW1P[lp]   = PACK(PR(L + 5), 256, 1024);
    gW2P[lp]   = PACK(PR(L + 6), 1024, 256);
    pWP[lp]    = PACK(PR(L + 11), 128, 256);
    aWoP[lp]   = PACK(PR(L + 15), 256, 64);
    aWqkvP[lp] = PACK(PR(L + 16), 64, 768);
    fW1P[lp]   = PACK(PR(L + 20), 64, 256);
    fW2P[lp]   = PACK(PR(L + 21), 256, 64);
  }

  // ---- geometry: transpose, FPS, kNN, input embeddings ----------------------
  k_xpose<<<(Bn * Np + 255) / 256, 256, 0, stream>>>(X, xyz, Bn, Np);
  k_fps<<<Bn, 256, 0, stream>>>(xyz, fidx, anch);
  k_knn<<<Bn * Pn, 256, 0, stream>>>(xyz, anch, aknn);
  k_edge<<<Bn * Pn * Kn, 64, 0, stream>>>(xyz, anch, aknn, Wpt, bpt, edges);
  k_anchor<<<Bn * Pn, 256, 0, stream>>>(anch, Wan, ban, p0);

  const float* patches = p0; int pld = 256;

  for (int lp = 0; lp < 4; ++lp) {
    int L = 18 + lp * 26;
    const float *gbo = PR(L + 2), *glnb = PR(L + 3), *glng = PR(L + 4);
    const float *gb1 = PR(L + 7), *gb2 = PR(L + 8), *gflnb = PR(L + 9), *gflng = PR(L + 10);
    const float *pb2p = PR(L + 12), *plnb = PR(L + 13), *plng = PR(L + 14);
    const float *abo = PR(L + 17), *alnb = PR(L + 18), *alng = PR(L + 19);
    const float *fb1 = PR(L + 22), *fb2 = PR(L + 23), *fflnb = PR(L + 24), *fflng = PR(L + 25);

    // ======== point-level transformer: 4096 seqs x 32 tokens x dim 64 ========
    const int CH = 1024;                       // seqs per chunk (keeps scratch small)
    for (int c0 = 0; c0 < Bn * Pn; c0 += CH) {
      float* ech = edges + (long)c0 * Kn * 64;
      int toks = CH * Kn;                      // 32768
      // attention: q,k stored f16 -> QK^T runs on pure f16 operands
      LNRM(ech, lnh, alng, alnb, toks, 64, 64, 64);
      LINH(lnh, aWqkvP[lp],            nullptr, nullptr, bufA, 1,
           toks, 64, 256, 64, 256, 0, 0);                                          // q (f16)
      LINH(lnh, aWqkvP[lp] + 256 * 64, nullptr, nullptr, bufB, 1,
           toks, 64, 256, 64, 256, 0, 0);                                          // k (f16)
      G_HH(bufAh, bufBh, nullptr, nullptr, nullptr, nullptr, Sbuf, 0,
           32, 64, 32, 256, 256, 32, 0,
           (long)Kn * 256, 64, (long)Kn * 256, 64, (long)HEADS * 1024, 1024, 0, 0,
           CH, HEADS, 0, SCALE);                                                   // S=qk^T
      SMAX(Sbuf, (long)CH * HEADS * 32, 32);
      LINH(lnh, aWqkvP[lp] + 512 * 64, nullptr, nullptr, bufB, 0,
           toks, 64, 256, 64, 256, 0, 0);                                          // v (fp32)
      G_FF(Sbuf, bufB, bufA, 1, 32, 32, 64, 32, 256, 256,
           (long)HEADS * 1024, 1024, (long)Kn * 256, 64, (long)Kn * 256, 64,
           CH, HEADS);                                                             // O=Pv (f16)
      LINH(bufAh, aWoP[lp], abo, ech, ech, 0, toks, 256, 64, 256, 64, 64, 0);      // proj+res
      // feed-forward: GELU hidden stored f16
      LNRM(ech, lnh, fflng, fflnb, toks, 64, 64, 64);
      LINH(lnh, fW1P[lp], fb1, nullptr, bufA, 1, toks, 64, 256, 64, 256, 0, 1);    // gelu f16
      LINH(bufAh, fW2P[lp], fb2, ech, ech, 0, toks, 256, 64, 256, 64, 64, 0);      // +res
    }

    // ======== point -> patch aggregation =====================================
    k_maxmean<<<Bn * Pn, 64, 0, stream>>>(edges, ef, Kn, 64, 64);                  // (4096,128)
    LNRM(ef, lnh, plng, plnb, Bn * Pn, 128, 128, 128);
    LINH(lnh, pWP[lp], pb2p, patches, pwork, 0,
         Bn * Pn, 128, 256, 128, 256, pld, 0);                                     // patches+=res

    // ======== patch-level transformer: 16 seqs x 256 tokens x dim 256 ========
    LNRM(pwork, lnh, glng, glnb, Bn * Pn, 256, 256, 256);
    LINH(lnh, gWqkvP[lp],             nullptr, nullptr, bufA, 1,
         Bn * Pn, 256, 256, 256, 256, 0, 0);                                       // q (f16)
    LINH(lnh, gWqkvP[lp] + 256 * 256, nullptr, nullptr, bufB, 1,
         Bn * Pn, 256, 256, 256, 256, 0, 0);                                       // k (f16)
    G_HH(bufAh, bufBh, nullptr, nullptr, nullptr, nullptr, Sbuf, 0,
         256, 64, 256, 256, 256, 256, 0,
         (long)Pn * 256, 64, (long)Pn * 256, 64, (long)HEADS * Pn * Pn, (long)Pn * Pn, 0, 0,
         Bn, HEADS, 0, SCALE);
    SMAX(Sbuf, (long)Bn * HEADS * Pn, Pn);
    LINH(lnh, gWqkvP[lp] + 512 * 256, nullptr, nullptr, bufB, 0,
         Bn * Pn, 256, 256, 256, 256, 0, 0);                                       // v (fp32)
    G_FF(Sbuf, bufB, bufA, 1, 256, 256, 64, 256, 256, 256,
         (long)HEADS * Pn * Pn, (long)Pn * Pn, (long)Pn * 256, 64, (long)Pn * 256, 64,
         Bn, HEADS);                                                               // O (f16)
    LINH(bufAh, gWoP[lp], gbo, pwork, pwork, 0,
         Bn * Pn, 256, 256, 256, 256, 256, 0);
    // FF; final add written straight into this layer's slice of the concat buf
    LNRM(pwork, lnh, gflng, gflnb, Bn * Pn, 256, 256, 256);
    LINH(lnh, gW1P[lp], gb1, nullptr, bufA, 1, Bn * Pn, 256, 1024, 256, 1024, 0, 1);
    LINH(bufAh, gW2P[lp], gb2, pwork, ybuf + lp * 256, 0,
         Bn * Pn, 1024, 256, 1024, 1024, 256, 0);
    patches = ybuf + lp * 256; pld = 1024;     // next layer reads from concat slice
  }

  // ---- final projection + pooling + BN-GELU head ----------------------------
  LNRM(ybuf, lnh, fclng, fclnb, Bn * Pn, 1024, 1024, 1024);
  LINH(lnh, fcWp, fcb, nullptr, bufA, 0, Bn * Pn, 1024, 1024, 1024, 1024, 0, 1);
  k_maxmean<<<Bn, 256, 0, stream>>>(bufA, pooled, Pn, 1024, 1024);                 // (16,2048)
  G_FH(pooled, hW1p, hb1, hg1, hbe1, h1b, 16, 2048, 512, 2048, 512, 1);            // bn+gelu
  G_FH(h1b, hW2p, hb2, hg2, hbe2, h2b, 16, 512, 256, 512, 256, 1);                 // bn+gelu
  G_FH(h2b, hW3p, hb3, nullptr, nullptr, (float*)d_out, 16, 256, 40, 256, 40, 0);  // logits
}
